// RNN_8744553414975
// MI455X (gfx1250) — matrix operand order repaired and checked by reading, not compiled, not run on hardware
//
#include <hip/hip_runtime.h>
#include <hip/hip_bf16.h>

// Problem constants
#define D     1024
#define BATCH 32
#define SEQ   512
#define KC    32      // number of K-chunks of 32 (D / 32)
#define NT    64      // number of 16-wide N tiles (D / 16)

typedef __bf16 bf16_t;
typedef __attribute__((ext_vector_type(16))) __bf16 bf16x16;
typedef __attribute__((ext_vector_type(8)))  __bf16 bf16x8;
typedef __attribute__((ext_vector_type(8)))  float  f32x8;

// ---------------------------------------------------------------------------
// Fragment loaders
// ---------------------------------------------------------------------------
// A-matrix 16x32 bf16, row-major source, lda = D.
// ISA layout: lanes 0-15 hold M=lane, K = {0..7, 16..23}; lanes 16-31 hold
// M=lane-16, K = {8..15, 24..31}. -> two contiguous 16B loads per lane.
static __device__ __forceinline__ bf16x16
load_a_frag(const bf16_t* __restrict__ A, int row_tile, int k0, int lane) {
    int r  = row_tile + (lane & 15);
    int kh = (lane >> 4) << 3;              // 0 or 8
    const bf16_t* p = A + (size_t)r * D + k0 + kh;
    bf16x8 lo = *(const bf16x8*)p;          // K = k0+kh .. +7
    bf16x8 hi = *(const bf16x8*)(p + 16);   // K = k0+16+kh .. +7
    bf16x16 a;
#pragma unroll
    for (int i = 0; i < 8; ++i) { a[i] = lo[i]; a[i + 8] = hi[i]; }
    return a;
}

// B fragments are pre-packed as [nt][kc][lane][16] bf16 -> one 32B load/lane.
static __device__ __forceinline__ bf16x16
load_b_frag(const bf16_t* __restrict__ P, int nt, int kc, int lane) {
    return *(const bf16x16*)(P + ((((size_t)nt * KC + kc) * 32) + lane) * 16);
}

static __device__ __forceinline__ f32x8
wmma_bf16(bf16x16 a, bf16x16 b, f32x8 c) {
    // (neg_a, A, neg_b, B, c_mod, C, reuse_a, reuse_b)
    return __builtin_amdgcn_wmma_f32_16x16x32_bf16(false, a, false, b,
                                                   (short)0, c, false, false);
}

// ---------------------------------------------------------------------------
// One-shot prep kernels
// ---------------------------------------------------------------------------
__global__ void k_f32_to_bf16(const float* __restrict__ src,
                              bf16_t* __restrict__ dst, int n) {
    int i = blockIdx.x * blockDim.x + threadIdx.x;
    if (i < n) dst[i] = (bf16_t)src[i];
}

__global__ void k_zero_bf16(bf16_t* __restrict__ p, int n) {
    int i = blockIdx.x * blockDim.x + threadIdx.x;
    if (i < n) p[i] = (bf16_t)0.0f;
}

// Pack the 4 weight matrices (wh0, uh0, wh1, uh1; each D x D, row = K, col = N)
// into WMMA B-fragment layout: dst[mat][nt][kc][lane][j] where
//   K = kc*32 + (lane>>4)*8 + (j&7) + (j>>3)*16,  N = nt*16 + (lane&15)
__global__ void k_pack_weights(const float* __restrict__ wh,
                               const float* __restrict__ uh,
                               bf16_t* __restrict__ dst) {
    const size_t per = (size_t)NT * KC * 32 * 16;   // 1,048,576 per matrix
    size_t e = (size_t)blockIdx.x * blockDim.x + threadIdx.x;
    if (e >= 4 * per) return;
    int    mat = (int)(e / per);
    size_t r   = e % per;
    int j    = (int)(r & 15);
    int lane = (int)((r >> 4)  & 31);
    int kc   = (int)((r >> 9)  & 31);
    int nt   = (int)(r >> 14);
    int K = kc * 32 + ((lane >> 4) << 3) + (j & 7) + ((j >> 3) << 4);
    int N = nt * 16 + (lane & 15);
    const float* src;
    switch (mat) {
        case 0:  src = wh;              break;   // wh[0]
        case 1:  src = uh;              break;   // uh[0]
        case 2:  src = wh + (size_t)D * D; break; // wh[1]
        default: src = uh + (size_t)D * D; break; // uh[1]
    }
    dst[e] = (bf16_t)src[(size_t)K * D + N];
}

// ---------------------------------------------------------------------------
// Parallel precompute: A0[t][b][:] = x[b][t][:] @ wh[0]   (f32 accumulate)
// Grid: 8192 blocks * 8 waves = 65536 waves = 1024 M-tiles * 64 N-tiles.
// ---------------------------------------------------------------------------
__global__ void __launch_bounds__(256)
k_precompute_wx(const bf16_t* __restrict__ xbf,
                const bf16_t* __restrict__ wh0p,
                float* __restrict__ A0) {
    int wid  = blockIdx.x * 8 + (threadIdx.x >> 5);
    int lane = threadIdx.x & 31;
    int mt = wid >> 6;            // 0..1023  (row tile over t*32+b)
    int nt = wid & 63;
    int t  = mt >> 1;
    int b0 = (mt & 1) << 4;       // 0 or 16
    int brow = b0 + (lane & 15);  // batch index of this lane's A row
    const bf16_t* arow = xbf + ((size_t)brow * SEQ + t) * D; // x is [B][S][D]
    int kh = (lane >> 4) << 3;
    f32x8 c = {};
#pragma unroll 4
    for (int kc = 0; kc < KC; ++kc) {
        const bf16_t* p = arow + kc * 32 + kh;
        bf16x8 lo = *(const bf16x8*)p;
        bf16x8 hi = *(const bf16x8*)(p + 16);
        bf16x16 a;
#pragma unroll
        for (int i = 0; i < 8; ++i) { a[i] = lo[i]; a[i + 8] = hi[i]; }
        bf16x16 b = load_b_frag(wh0p, nt, kc, lane);
        c = wmma_bf16(a, b, c);
    }
    int col   = nt * 16 + (lane & 15);
    int rbase = b0 + ((lane >> 4) << 3);     // row within the 32-batch slab
#pragma unroll
    for (int i = 0; i < 8; ++i)
        A0[((size_t)t * BATCH + rbase + i) * D + col] = c[i];
}

// ---------------------------------------------------------------------------
// Pipelined recurrent step. Launch t = 0..SEQ (inclusive):
//   blocks  0..15 : layer0(t)    h0(t)   = tanh(A0[t] + h0(t-1) @ uh0 + b0)
//   blocks 16..31 : layer1(t-1)  h1(t-1) = tanh(h0(t-1)@wh1 + h1(t-2)@uh1 + b1)
// h0 / h1 double-buffered by step parity (state s lives in buf[s&1]).
// ---------------------------------------------------------------------------
__global__ void __launch_bounds__(256)
k_rnn_step(int t,
           const float*  __restrict__ A0,
           const bf16_t* __restrict__ uh0p,
           const bf16_t* __restrict__ wh1p,
           const bf16_t* __restrict__ uh1p,
           bf16_t* __restrict__ h0a, bf16_t* __restrict__ h0b,
           bf16_t* __restrict__ h1a, bf16_t* __restrict__ h1b,
           const float* __restrict__ bias,
           float* __restrict__ out) {
    const int half = blockIdx.x >> 4;
    const int wid  = (blockIdx.x & 15) * 8 + (threadIdx.x >> 5);
    const int lane = threadIdx.x & 31;
    const int mt   = wid >> 6;                    // 0..1
    const int nt   = wid & 63;
    const int col  = nt * 16 + (lane & 15);
    const int rbase = mt * 16 + ((lane >> 4) << 3);

    if (half == 0) {                              // ---- layer 0, step t ----
        if (t >= SEQ) return;
        const bf16_t* h0_prev = (t & 1) ? h0a : h0b;  // h0(t-1) in buf[(t-1)&1]
        bf16_t*       h0_cur  = (t & 1) ? h0b : h0a;  // h0(t)   in buf[t&1]
        f32x8 c;
#pragma unroll
        for (int i = 0; i < 8; ++i)               // C init = precomputed x@wh0
            c[i] = A0[((size_t)t * BATCH + rbase + i) * D + col];
#pragma unroll 4
        for (int kc = 0; kc < KC; ++kc) {
            bf16x16 a = load_a_frag(h0_prev, mt * 16, kc * 32, lane);
            bf16x16 b = load_b_frag(uh0p, nt, kc, lane);
            c = wmma_bf16(a, b, c);
        }
        float bv = bias[col];
#pragma unroll
        for (int i = 0; i < 8; ++i) {
            float v = tanhf(c[i] + bv);
            h0_cur[(size_t)(rbase + i) * D + col] = (bf16_t)v;
        }
    } else {                                      // ---- layer 1, step t-1 ----
        const int s = t - 1;
        if (s < 0) return;
        const bf16_t* h0_s     = (s & 1) ? h0b : h0a; // h0(s)
        const bf16_t* h1_prev  = (s & 1) ? h1a : h1b; // h1(s-1) in buf[(s-1)&1]
        bf16_t*       h1_cur   = (s & 1) ? h1b : h1a; // h1(s)   in buf[s&1]
        f32x8 c = {};
#pragma unroll 2
        for (int kc = 0; kc < KC; ++kc) {
            bf16x16 aw = load_a_frag(h0_s, mt * 16, kc * 32, lane);
            bf16x16 bw = load_b_frag(wh1p, nt, kc, lane);
            c = wmma_bf16(aw, bw, c);
            bf16x16 au = load_a_frag(h1_prev, mt * 16, kc * 32, lane);
            bf16x16 bu = load_b_frag(uh1p, nt, kc, lane);
            c = wmma_bf16(au, bu, c);
        }
        float bv = bias[D + col];
#pragma unroll
        for (int i = 0; i < 8; ++i) {
            int b = rbase + i;                    // batch index
            float v = tanhf(c[i] + bv);
            h1_cur[(size_t)b * D + col] = (bf16_t)v;
            out[((size_t)b * SEQ + s) * D + col] = v;   // out[B][S][D]
        }
    }
}

// ---------------------------------------------------------------------------
// Host-side launch sequence (graph-capture safe: kernels on `stream` only).
// ---------------------------------------------------------------------------
extern "C" void kernel_launch(void* const* d_in, const int* in_sizes, int n_in,
                              void* d_out, int out_size, void* d_ws, size_t ws_size,
                              hipStream_t stream) {
    const float* x    = (const float*)d_in[0];   // [32][512][1024]
    const float* wh   = (const float*)d_in[1];   // [2][1024][1024]
    const float* uh   = (const float*)d_in[2];   // [2][1024][1024]
    const float* bias = (const float*)d_in[3];   // [2][1024]
    float* out = (float*)d_out;

    char* ws = (char*)d_ws;
    const size_t XBF_BYTES  = (size_t)BATCH * SEQ * D * 2;        // 32 MB
    const size_t PACK_ELEMS = (size_t)NT * KC * 32 * 16;          // 1 Mi per matrix
    const size_t PACK_BYTES = 4 * PACK_ELEMS * 2;                 // 8 MB
    const size_t A0_BYTES   = (size_t)SEQ * BATCH * D * 4;        // 64 MB

    bf16_t* xbf    = (bf16_t*)ws;
    bf16_t* packed = (bf16_t*)(ws + XBF_BYTES);
    float*  A0     = (float*)(ws + XBF_BYTES + PACK_BYTES);
    bf16_t* h0a    = (bf16_t*)(ws + XBF_BYTES + PACK_BYTES + A0_BYTES);
    bf16_t* h0b    = h0a + (size_t)BATCH * D;
    bf16_t* h1a    = h0b + (size_t)BATCH * D;
    bf16_t* h1b    = h1a + (size_t)BATCH * D;

    const bf16_t* wh0p = packed + 0 * PACK_ELEMS;
    const bf16_t* uh0p = packed + 1 * PACK_ELEMS;
    const bf16_t* wh1p = packed + 2 * PACK_ELEMS;
    const bf16_t* uh1p = packed + 3 * PACK_ELEMS;

    // 1) x -> bf16
    {
        int n = BATCH * SEQ * D;                               // 16,777,216
        k_f32_to_bf16<<<n / 256, 256, 0, stream>>>(x, xbf, n);
    }
    // 2) pack weights into WMMA B layout (bf16)
    {
        int n = (int)(4 * PACK_ELEMS);                         // 4,194,304
        k_pack_weights<<<n / 256, 256, 0, stream>>>(wh, uh, packed);
    }
    // 3) zero recurrent state (h0a..h1b are contiguous)
    {
        int n = 4 * BATCH * D;                                 // 131,072
        k_zero_bf16<<<n / 256, 256, 0, stream>>>(h0a, n);
    }
    // 4) parallel precompute of x @ wh[0] for all timesteps
    k_precompute_wx<<<8192, 256, 0, stream>>>(xbf, wh0p, A0);

    // 5) sequential recurrence, two layers pipelined per launch
    for (int t = 0; t <= SEQ; ++t)
        k_rnn_step<<<32, 256, 0, stream>>>(t, A0, uh0p, wh1p, uh1p,
                                           h0a, h0b, h1a, h1b, bias, out);
}
